// LSTM_59691455480484
// MI455X (gfx1250) — matrix operand order repaired and checked by reading, not compiled, not run on hardware
//
#include <hip/hip_runtime.h>
#include <stdint.h>

#define TT 1024
#define BB 32
#define DD 512
#define HH 512
#define GH 2048            // 4*H
#define KTOT 1024          // D + H (same for both layers)
#define NWG 32             // workgroups in recurrent kernel
#define WPB 2              // waves per block (wave32)

typedef __bf16 bf16_t;
typedef bf16_t v16bf __attribute__((ext_vector_type(16)));
typedef float  v8f   __attribute__((ext_vector_type(8)));

// ---------- helpers ----------

__device__ __forceinline__ unsigned short f2bf(float f) {
  union { float f; unsigned u; } v; v.f = f;
  unsigned u = v.u;
  if ((u & 0x7f800000u) != 0x7f800000u)          // not inf/nan: round to nearest even
    u = u + 0x7fffu + ((u >> 16) & 1u);
  return (unsigned short)(u >> 16);
}

struct TwoU4 { uint4 a, b; };

// Build a 32-byte WMMA fragment from two contiguous 16-byte runs.
__device__ __forceinline__ v16bf load_frag2(const unsigned short* p0,
                                            const unsigned short* p1) {
  TwoU4 t;
  t.a = *(const uint4*)p0;
  t.b = *(const uint4*)p1;
  return __builtin_bit_cast(v16bf, t);
}

__device__ __forceinline__ float sigf(float x) {
  return 1.0f / (1.0f + __expf(-x));
}

// ---------- prep kernels ----------

__global__ void k_f32_to_bf16(const float* __restrict__ src,
                              unsigned short* __restrict__ dst, size_t n) {
  size_t i = (size_t)blockIdx.x * blockDim.x + threadIdx.x;
  size_t stride = (size_t)gridDim.x * blockDim.x;
  for (; i < n; i += stride) dst[i] = f2bf(src[i]);
}

// wT[n][k] = bf16( k<D ? w_ih[k][n] : w_hh[k-D][n] )   (N-major, K contiguous)
__global__ void k_prep_wT(const float* __restrict__ w_ih,
                          const float* __restrict__ w_hh,
                          unsigned short* __restrict__ wT) {
  size_t idx = (size_t)blockIdx.x * blockDim.x + threadIdx.x;
  if (idx >= (size_t)GH * KTOT) return;
  int n = (int)(idx >> 10);
  int k = (int)(idx & 1023);
  float v = (k < DD) ? w_ih[(size_t)k * GH + n]
                     : w_hh[(size_t)(k - DD) * GH + n];
  wT[idx] = f2bf(v);
}

__global__ void k_prep_misc(const float* __restrict__ b_i0, const float* __restrict__ b_h0,
                            const float* __restrict__ b_i1, const float* __restrict__ b_h1,
                            const float* __restrict__ h0,   // [L,B,H]
                            float* __restrict__ bsum0, float* __restrict__ bsum1,
                            unsigned short* __restrict__ hb0,
                            unsigned short* __restrict__ hb1,
                            unsigned int* __restrict__ ctrs) {
  int i = blockIdx.x * blockDim.x + threadIdx.x;
  if (i < GH) { bsum0[i] = b_i0[i] + b_h0[i]; bsum1[i] = b_i1[i] + b_h1[i]; }
  if (i < BB * HH) {
    hb0[i] = f2bf(h0[i]);               // ping buffer 0 of layer 0
    hb1[i] = f2bf(h0[BB * HH + i]);     // ping buffer 0 of layer 1
  }
  if (i < 2) ctrs[i] = 0u;
}

// ---------- persistent recurrent LSTM layer ----------
//
// 64 waves total: wave (m, j) owns batch-tile m (16 rows) x hidden-tile j (16 cols)
// and computes all four gate tiles f,i,o,g so the cell state stays in registers.

__global__ __launch_bounds__(WPB * 32, 1)
void lstm_layer_kernel(const unsigned short* __restrict__ xseq, // [T,B,512] bf16
                       const unsigned short* __restrict__ wT,   // [2048,1024] bf16 (N-major)
                       const float* __restrict__ bsum,          // [2048] = b_i + b_h
                       const float* __restrict__ h0,            // [B,H] layer slice
                       const float* __restrict__ c0,            // [B,H] layer slice
                       const int*   __restrict__ length,        // [B]
                       unsigned short* __restrict__ hbuf,       // [2][B,512] bf16 ping-pong
                       unsigned short* __restrict__ seqbf,      // [T,B,512] bf16 out or null
                       float* __restrict__ seqf,                // [T,B,512] f32 out or null
                       float* __restrict__ hn,                  // [B,H]
                       float* __restrict__ cn,                  // [B,H]
                       unsigned int* __restrict__ ctr)
{
  const int lane = threadIdx.x & 31;
  const int wid  = threadIdx.x >> 5;
  const int gw   = blockIdx.x * WPB + wid;   // 0..63
  const int m    = gw & 1;                   // batch tile (0/1)
  const int j    = gw >> 1;                  // hidden tile (0..31)
  const int nlo  = lane & 15;
  const int hi   = lane >> 4;
  const int hid  = j * 16 + nlo;             // hidden column this lane holds in C/D
  const int arow = m * 16 + nlo;             // A-matrix row this lane serves

  float c_reg[8], h_reg[8], bias[4];
  int   len_e[8];
  for (int e = 0; e < 8; ++e) {
    int batch = m * 16 + e + 8 * hi;         // C/D row mapping: VGPR e, lane-half hi
    c_reg[e] = c0[(size_t)batch * HH + hid];
    h_reg[e] = h0[(size_t)batch * HH + hid];
    len_e[e] = length[batch];
  }
  for (int g = 0; g < 4; ++g) bias[g] = bsum[g * HH + hid];

  for (int t = 0; t < TT; ++t) {
    v8f acc[4];
    for (int g = 0; g < 4; ++g)
      for (int e = 0; e < 8; ++e) acc[g][e] = 0.0f;

    const unsigned short* xrow = xseq + ((size_t)t * BB + arow) * DD;
    const unsigned short* hrow = hbuf + (size_t)(t & 1) * BB * HH + (size_t)arow * HH;

    // K = 1024 = [x_t (512) | h_{t-1} (512)], 32 K-blocks of 32
    #pragma unroll 4
    for (int kb = 0; kb < KTOT / 32; ++kb) {
      int k0 = kb * 32;
      const unsigned short* ar = (k0 < DD) ? (xrow + k0) : (hrow + (k0 - DD));
      // A 16x32 bf16, k order chosen here (matches B): lane<16 K0..15, lane>=16 K16..31
      v16bf a = load_frag2(ar + hi * 16, ar + hi * 16 + 8);
      for (int g = 0; g < 4; ++g) {
        // B 32x16 bf16, k order chosen here (matches A): col = lane%16; lane<16 K0..15, lane>=16 K16..31
        const unsigned short* bp =
            wT + (size_t)(g * HH + hid) * KTOT + k0 + hi * 16;
        v16bf b = load_frag2(bp, bp + 8);
        acc[g] = __builtin_amdgcn_wmma_f32_16x16x32_bf16(
            false, a, false, b, (short)0, acc[g], false, false);
      }
    }

    // pointwise LSTM cell (f,i,o,g order) with length masking
    unsigned short* hnext = hbuf + (size_t)((t + 1) & 1) * BB * HH;
    for (int e = 0; e < 8; ++e) {
      int batch = m * 16 + e + 8 * hi;
      float fs = sigf(acc[0][e] + bias[0]);
      float is = sigf(acc[1][e] + bias[1]);
      float os = sigf(acc[2][e] + bias[2]);
      float gt = tanhf(acc[3][e] + bias[3]);
      float c1 = fs * c_reg[e] + is * gt;
      float h1 = os * tanhf(c1);
      bool upd = (t < len_e[e]);
      c_reg[e] = upd ? c1 : c_reg[e];
      h_reg[e] = upd ? h1 : h_reg[e];
      size_t off = (size_t)batch * HH + hid;
      hnext[off] = f2bf(h_reg[e]);
      if (seqbf) seqbf[(size_t)t * BB * HH + off] = f2bf(h_reg[e]);
      if (seqf)  seqf [(size_t)t * BB * HH + off] = h_reg[e];
    }

    // device-wide step barrier (monotonic counter; zeroed by prep each launch)
    __threadfence();
    __syncthreads();
    if (threadIdx.x == 0) {
      __hip_atomic_fetch_add(ctr, 1u, __ATOMIC_RELEASE, __HIP_MEMORY_SCOPE_AGENT);
      unsigned int target = (unsigned int)NWG * (unsigned int)(t + 1);
      while (__hip_atomic_load(ctr, __ATOMIC_ACQUIRE, __HIP_MEMORY_SCOPE_AGENT) < target)
        __builtin_amdgcn_s_sleep(2);
    }
    __syncthreads();
    __threadfence();
  }

  // final states (full precision, post-mask)
  for (int e = 0; e < 8; ++e) {
    int batch = m * 16 + e + 8 * hi;
    hn[(size_t)batch * HH + hid] = h_reg[e];
    cn[(size_t)batch * HH + hid] = c_reg[e];
  }
}

// ---------- launcher ----------

extern "C" void kernel_launch(void* const* d_in, const int* in_sizes, int n_in,
                              void* d_out, int out_size, void* d_ws, size_t ws_size,
                              hipStream_t stream) {
  (void)in_sizes; (void)n_in; (void)out_size; (void)ws_size;

  const float* x    = (const float*)d_in[0];
  const float* h0   = (const float*)d_in[1];
  const float* c0   = (const float*)d_in[2];
  const int*   len  = (const int*)  d_in[3];
  const float* wih0 = (const float*)d_in[4];
  const float* whh0 = (const float*)d_in[5];
  const float* bi0  = (const float*)d_in[6];
  const float* bh0  = (const float*)d_in[7];
  const float* wih1 = (const float*)d_in[8];
  const float* whh1 = (const float*)d_in[9];
  const float* bi1  = (const float*)d_in[10];
  const float* bh1  = (const float*)d_in[11];
  float* out = (float*)d_out;

  // workspace layout (~76 MB)
  unsigned short* xbf  = (unsigned short*)d_ws;            // T*B*D bf16
  unsigned short* seq0 = xbf  + (size_t)TT * BB * DD;      // T*B*H bf16 (layer0 out)
  unsigned short* wT0  = seq0 + (size_t)TT * BB * HH;      // 2048*1024 bf16
  unsigned short* wT1  = wT0  + (size_t)GH * KTOT;
  unsigned short* hb0  = wT1  + (size_t)GH * KTOT;         // 2*B*H bf16 ping-pong
  unsigned short* hb1  = hb0  + 2 * (size_t)BB * HH;
  float* bsum0 = (float*)(hb1 + 2 * (size_t)BB * HH);
  float* bsum1 = bsum0 + GH;
  unsigned int* ctrs = (unsigned int*)(bsum1 + GH);

  float* xout = out;                                  // [T,B,H]
  float* hn   = out + (size_t)TT * BB * HH;           // [L,B,H]
  float* cn   = hn + 2 * (size_t)BB * HH;             // [L,B,H]

  k_prep_misc<<<64, 256, 0, stream>>>(bi0, bh0, bi1, bh1, h0,
                                      bsum0, bsum1, hb0, hb1, ctrs);
  k_f32_to_bf16<<<4096, 256, 0, stream>>>(x, xbf, (size_t)TT * BB * DD);
  {
    int blocks = (int)(((size_t)GH * KTOT + 255) / 256);
    k_prep_wT<<<blocks, 256, 0, stream>>>(wih0, whh0, wT0);
    k_prep_wT<<<blocks, 256, 0, stream>>>(wih1, whh1, wT1);
  }

  lstm_layer_kernel<<<NWG, WPB * 32, 0, stream>>>(
      xbf, wT0, bsum0, h0, c0, len,
      hb0, seq0, nullptr, hn, cn, ctrs + 0);

  lstm_layer_kernel<<<NWG, WPB * 32, 0, stream>>>(
      seq0, wT1, bsum1, h0 + (size_t)BB * HH, c0 + (size_t)BB * HH, len,
      hb1, nullptr, xout, hn + (size_t)BB * HH, cn + (size_t)BB * HH, ctrs + 1);
}
